// Speller_30374008717418
// MI455X (gfx1250) — matrix operand order repaired and checked by reading, not compiled, not run on hardware
//
#include <hip/hip_runtime.h>

// ---------------------------------------------------------------------------
// LAS "Speller" decoder for MI455X (gfx1250, wave32, WMMA).
//
// Strategy (compile-only; reasoned from MI455X specs):
//  * One-time init kernels convert weights -> bf16 (W_ih|W_hh fused to a single
//    [2048][1056] matrix), listener features -> bf16 (67MB, L2-resident on the
//    192MB L2), biases fused.
//  * Per step: gates GEMM via v_wmma_f32_16x16x32_bf16 (M=64,N=2048,K=1056),
//    LSTM pointwise, attention energy matvec (bandwidth-bound -> VALU with
//    coalesced bf16 loads + wave32 shfl reduction), softmax+context (writes
//    att_seq), output GEMM via WMMA (M=64,N=32,K=1024) + fused log_softmax.
// ---------------------------------------------------------------------------

#define BB 64
#define TT 1024
#define DD 512
#define VV 32
#define LL 100
#define KX 1056   /* V + D + D : [word | ctx | h] */
#define NG 2048   /* 4*D gate outputs */

typedef __bf16 bf16_t;
typedef __attribute__((ext_vector_type(16))) bf16_t v16bf;
typedef __attribute__((ext_vector_type(8)))  bf16_t v8bf;
typedef __attribute__((ext_vector_type(8)))  float  v8f;

__device__ __forceinline__ bf16_t f2bf(float f) {
  union { float f; unsigned u; } v; v.f = f;
  unsigned r = v.u + 0x7FFFu + ((v.u >> 16) & 1u);   // round-to-nearest-even
  unsigned short h = (unsigned short)(r >> 16);
  return __builtin_bit_cast(bf16_t, h);
}
__device__ __forceinline__ float bf2f(bf16_t b) {
  unsigned short h = __builtin_bit_cast(unsigned short, b);
  union { unsigned u; float f; } v; v.u = ((unsigned)h) << 16;
  return v.f;
}
__device__ __forceinline__ float sigmoidf_(float x) { return 1.0f / (1.0f + __expf(-x)); }

// ------------------------------- init kernels ------------------------------

__global__ void k_init_wg(const float* __restrict__ Wih, const float* __restrict__ Whh,
                          bf16_t* __restrict__ Wg) {
  int idx = blockIdx.x * blockDim.x + threadIdx.x;       // over 2048*1056
  if (idx >= NG * KX) return;
  int n = idx / KX, k = idx % KX;
  float v = (k < (VV + DD)) ? Wih[n * (VV + DD) + k] : Whh[n * DD + (k - (VV + DD))];
  Wg[idx] = f2bf(v);
}

__global__ void k_init_bg(const float* __restrict__ bih, const float* __restrict__ bhh,
                          float* __restrict__ bg) {
  int n = blockIdx.x * blockDim.x + threadIdx.x;
  if (n < NG) bg[n] = bih[n] + bhh[n];
}

__global__ void k_init_wout(const float* __restrict__ Wout, bf16_t* __restrict__ Wob) {
  int idx = blockIdx.x * blockDim.x + threadIdx.x;       // over 32*1024
  if (idx < VV * 2 * DD) Wob[idx] = f2bf(Wout[idx]);
}

__global__ void k_init_lf(const float* __restrict__ lf, bf16_t* __restrict__ lfb) {
  size_t idx = (size_t)blockIdx.x * blockDim.x + threadIdx.x;  // over 64*1024*512
  lfb[idx] = f2bf(lf[idx]);
}

__global__ void k_init_state(const float* __restrict__ lf,
                             float* __restrict__ h, float* __restrict__ c,
                             bf16_t* __restrict__ xh) {
  int idx = blockIdx.x * blockDim.x + threadIdx.x;       // over 64*512
  int b = idx >> 9, d = idx & (DD - 1);
  h[idx] = 0.0f;
  c[idx] = 0.0f;
  xh[b * KX + VV + DD + d] = f2bf(0.0f);                  // h slot
  xh[b * KX + VV + d] = f2bf(lf[((size_t)b * TT + 0) * DD + d]);  // ctx0 = lf[:,0,:]
}

// ------------------------- gates GEMM (WMMA, bf16) -------------------------
// gates[64][2048] = [word|ctx|h] (64 x 1056) @ [Wih|Whh]^T.  One wave per
// 16x16 output tile; B fragment = one contiguous 32B row chunk of Wg[N][K].

__global__ void k_gates_gemm(const bf16_t* __restrict__ Wg, const bf16_t* __restrict__ xh,
                             const float* __restrict__ gt, int step,
                             float* __restrict__ gates) {
  const int wave = threadIdx.x >> 5;
  const int lane = threadIdx.x & 31;
  const int tile = blockIdx.x * (blockDim.x >> 5) + wave;  // 4*128 = 512 tiles
  const int mt = tile >> 7, nt = tile & 127;
  const int m0 = mt * 16, n0 = nt * 16;
  const int g = lane >> 4, lr = lane & 15;
  const int m = m0 + lr;                                   // A row (batch)
  const int n = n0 + lr;                                   // B row (gate unit)

  v8f acc;
#pragma unroll
  for (int i = 0; i < 8; ++i) acc[i] = 0.0f;

  // ---- k-tile 0: word one-hot (l==0) or ground_truth[:, step-1, :] ----
  {
    v16bf a;
#pragma unroll
    for (int e = 0; e < 16; ++e) {
      int kk = g * 16 + e;
      float w = (step == 0) ? ((kk == 0) ? 1.0f : 0.0f)
                            : gt[(size_t)m * (LL * VV) + (size_t)(step - 1) * VV + kk];
      a[e] = f2bf(w);
    }
    v16bf bm = *(const v16bf*)(Wg + (size_t)n * KX + g * 16);
    acc = __builtin_amdgcn_wmma_f32_16x16x32_bf16(false, a, false, bm,
                                                  (short)0, acc, false, false);
  }
  // ---- k-tiles 1..32: ctx + h from xh buffer ----
  for (int kt = 1; kt < 33; ++kt) {
    const bf16_t* ap = xh + (size_t)m * KX + kt * 32 + g * 16;
    v8bf lo = *(const v8bf*)(ap);
    v8bf hi = *(const v8bf*)(ap + 8);
    v16bf a;
#pragma unroll
    for (int e = 0; e < 8; ++e) { a[e] = lo[e]; a[8 + e] = hi[e]; }
    v16bf bm = *(const v16bf*)(Wg + (size_t)n * KX + kt * 32 + g * 16);
    acc = __builtin_amdgcn_wmma_f32_16x16x32_bf16(false, a, false, bm,
                                                  (short)0, acc, false, false);
  }
  // D layout: lane -> N=lr, VGPR r -> M = r + 8*g
#pragma unroll
  for (int r = 0; r < 8; ++r)
    gates[(size_t)(m0 + g * 8 + r) * NG + n0 + lr] = acc[r];
}

// ------------------------------ LSTM pointwise -----------------------------

__global__ void k_lstm(const float* __restrict__ gates, const float* __restrict__ bg,
                       float* __restrict__ h, float* __restrict__ c,
                       bf16_t* __restrict__ xh, bf16_t* __restrict__ concat) {
  int idx = blockIdx.x * blockDim.x + threadIdx.x;        // over 64*512
  int b = idx >> 9, d = idx & (DD - 1);
  const float* gr = gates + (size_t)b * NG;
  float i_ = sigmoidf_(gr[d]            + bg[d]);
  float f_ = sigmoidf_(gr[DD + d]       + bg[DD + d]);
  float g_ = tanhf    (gr[2 * DD + d]   + bg[2 * DD + d]);
  float o_ = sigmoidf_(gr[3 * DD + d]   + bg[3 * DD + d]);
  float cn = f_ * c[idx] + i_ * g_;
  float hn = o_ * tanhf(cn);
  c[idx] = cn;
  h[idx] = hn;
  xh[b * KX + VV + DD + d] = f2bf(hn);                    // next-step gates input
  concat[b * 2 * DD + d]   = f2bf(hn);                    // output-GEMM A, h half
}

// -------------------- attention energy: wave-per-(b,t) ---------------------
// energy[b][t] = sum_d h[b][d] * lf[b][t][d].  Bandwidth-bound matvec:
// 32 lanes stream coalesced 16B bf16 chunks, wave32 shfl_xor reduction.

__global__ void k_energy(const bf16_t* __restrict__ lfb, const float* __restrict__ h,
                         float* __restrict__ energy) {
  const int wave = threadIdx.x >> 5;
  const int lane = threadIdx.x & 31;
  const int t = blockIdx.x * (blockDim.x >> 5) + wave;
  const int b = blockIdx.y;
  const bf16_t* row = lfb + ((size_t)b * TT + t) * DD;
  const float*  hb  = h + (size_t)b * DD;
  float acc = 0.0f;
#pragma unroll
  for (int j = 0; j < 2; ++j) {
    int d0 = j * 256 + lane * 8;
    v8bf x = *(const v8bf*)(row + d0);
    v8f  hv = *(const v8f*)(hb + d0);
#pragma unroll
    for (int e = 0; e < 8; ++e) acc += bf2f(x[e]) * hv[e];
  }
#pragma unroll
  for (int off = 16; off; off >>= 1) acc += __shfl_xor(acc, off, 32);
  if (lane == 0) energy[(size_t)b * TT + t] = acc;
}

// ----------------- softmax over T + context matvec (fused) -----------------

__global__ void k_softmax_ctx(const float* __restrict__ energy, const bf16_t* __restrict__ lfb,
                              bf16_t* __restrict__ xh, bf16_t* __restrict__ concat,
                              float* __restrict__ att_out) {
  __shared__ float sc[TT];
  __shared__ float red[8];
  const int b = blockIdx.x;
  const int tid = threadIdx.x;
  const int wave = tid >> 5, lane = tid & 31;

  float lmax = -3.4e38f;
#pragma unroll
  for (int i = 0; i < 4; ++i) {
    int t = tid + i * 256;
    float e = energy[(size_t)b * TT + t];
    sc[t] = e;
    lmax = fmaxf(lmax, e);
  }
#pragma unroll
  for (int off = 16; off; off >>= 1) lmax = fmaxf(lmax, __shfl_xor(lmax, off, 32));
  if (lane == 0) red[wave] = lmax;
  __syncthreads();
  float m = red[0];
#pragma unroll
  for (int w = 1; w < 8; ++w) m = fmaxf(m, red[w]);
  __syncthreads();

  float lsum = 0.0f;
#pragma unroll
  for (int i = 0; i < 4; ++i) {
    int t = tid + i * 256;
    float ex = __expf(sc[t] - m);
    sc[t] = ex;
    lsum += ex;
  }
#pragma unroll
  for (int off = 16; off; off >>= 1) lsum += __shfl_xor(lsum, off, 32);
  if (lane == 0) red[wave] = lsum;
  __syncthreads();
  float s = red[0];
#pragma unroll
  for (int w = 1; w < 8; ++w) s += red[w];
  float inv = 1.0f / s;

#pragma unroll
  for (int i = 0; i < 4; ++i) {
    int t = tid + i * 256;
    float v = sc[t] * inv;
    sc[t] = v;
    att_out[(size_t)b * TT + t] = v;                      // att_seq[l][b][t]
  }
  __syncthreads();

  // ctx[b][d] = sum_t score[t] * lf[b][t][d]; coalesced along d.
  const bf16_t* base = lfb + (size_t)b * TT * DD;
  const int d0 = tid, d1 = tid + 256;
  float a0 = 0.0f, a1 = 0.0f;
  for (int t = 0; t < TT; ++t) {
    float sv = sc[t];
    const bf16_t* r = base + (size_t)t * DD;
    a0 += sv * bf2f(r[d0]);
    a1 += sv * bf2f(r[d1]);
  }
  xh[b * KX + VV + d0] = f2bf(a0);
  xh[b * KX + VV + d1] = f2bf(a1);
  concat[b * 2 * DD + DD + d0] = f2bf(a0);
  concat[b * 2 * DD + DD + d1] = f2bf(a1);
}

// ------------- output GEMM (WMMA) + fused log_softmax over V=32 ------------
// One block, 8 waves = 8 tiles (M=64 -> 4 tiles, N=32 -> 2 tiles), K=1024.

__global__ void k_out_gemm(const bf16_t* __restrict__ concat, const bf16_t* __restrict__ Wob,
                           const float* __restrict__ bout, float* __restrict__ raw_out) {
  __shared__ float raw[BB * VV];
  const int wave = threadIdx.x >> 5;
  const int lane = threadIdx.x & 31;
  const int mt = wave >> 1, nt = wave & 1;
  const int m0 = mt * 16, n0 = nt * 16;
  const int g = lane >> 4, lr = lane & 15;
  const int m = m0 + lr, n = n0 + lr;

  v8f acc;
#pragma unroll
  for (int i = 0; i < 8; ++i) acc[i] = 0.0f;
  for (int kt = 0; kt < 32; ++kt) {
    const bf16_t* ap = concat + (size_t)m * (2 * DD) + kt * 32 + g * 16;
    v8bf lo = *(const v8bf*)(ap);
    v8bf hi = *(const v8bf*)(ap + 8);
    v16bf a;
#pragma unroll
    for (int e = 0; e < 8; ++e) { a[e] = lo[e]; a[8 + e] = hi[e]; }
    v16bf bm = *(const v16bf*)(Wob + (size_t)n * (2 * DD) + kt * 32 + g * 16);
    acc = __builtin_amdgcn_wmma_f32_16x16x32_bf16(false, a, false, bm,
                                                  (short)0, acc, false, false);
  }
#pragma unroll
  for (int r = 0; r < 8; ++r)
    raw[(m0 + g * 8 + r) * VV + n0 + lr] = acc[r] + bout[n0 + lr];
  __syncthreads();

  if (threadIdx.x < BB) {
    const float* rr = raw + threadIdx.x * VV;
    float mx = rr[0];
#pragma unroll
    for (int v = 1; v < VV; ++v) mx = fmaxf(mx, rr[v]);
    float sum = 0.0f;
#pragma unroll
    for (int v = 0; v < VV; ++v) sum += __expf(rr[v] - mx);
    float lse = mx + __logf(sum);
#pragma unroll
    for (int v = 0; v < VV; ++v)
      raw_out[(size_t)threadIdx.x * VV + v] = rr[v] - lse;  // log_softmax
  }
}

// --------------------------------- launch ----------------------------------

extern "C" void kernel_launch(void* const* d_in, const int* in_sizes, int n_in,
                              void* d_out, int out_size, void* d_ws, size_t ws_size,
                              hipStream_t stream) {
  (void)in_sizes; (void)n_in; (void)out_size; (void)ws_size;
  const float* lf   = (const float*)d_in[0];
  const float* gt   = (const float*)d_in[1];
  const float* Wih  = (const float*)d_in[2];
  const float* Whh  = (const float*)d_in[3];
  const float* bih  = (const float*)d_in[4];
  const float* bhh  = (const float*)d_in[5];
  const float* Wout = (const float*)d_in[6];
  const float* bout = (const float*)d_in[7];
  float* out = (float*)d_out;

  char* ws = (char*)d_ws;
  size_t off = 0;
  auto carve = [&](size_t bytes) -> void* {
    void* p = ws + off;
    off = (off + bytes + 255) & ~(size_t)255;
    return p;
  };
  bf16_t* Wg     = (bf16_t*)carve((size_t)NG * KX * 2);          // 4.3 MB
  float*  bg     = (float*) carve((size_t)NG * 4);
  bf16_t* Wob    = (bf16_t*)carve((size_t)VV * 2 * DD * 2);
  bf16_t* lfb    = (bf16_t*)carve((size_t)BB * TT * DD * 2);     // 67 MB (L2-resident)
  bf16_t* xh     = (bf16_t*)carve((size_t)BB * KX * 2);
  bf16_t* concat = (bf16_t*)carve((size_t)BB * 2 * DD * 2);
  float*  hbuf   = (float*) carve((size_t)BB * DD * 4);
  float*  cbuf   = (float*) carve((size_t)BB * DD * 4);
  float*  gates  = (float*) carve((size_t)BB * NG * 4);
  float*  energy = (float*) carve((size_t)BB * TT * 4);

  float* raw_base = out;                                  // (L,B,V)
  float* att_base = out + (size_t)LL * BB * VV;           // (L,B,T)

  // ---- one-time init (re-run every call: pure function of inputs) ----
  k_init_wg   <<<(NG * KX + 255) / 256, 256, 0, stream>>>(Wih, Whh, Wg);
  k_init_bg   <<<(NG + 255) / 256,       256, 0, stream>>>(bih, bhh, bg);
  k_init_wout <<<(VV * 2 * DD + 255) / 256, 256, 0, stream>>>(Wout, Wob);
  k_init_lf   <<<(BB * TT * DD) / 256,   256, 0, stream>>>(lf, lfb);
  k_init_state<<<(BB * DD) / 256,        256, 0, stream>>>(lf, hbuf, cbuf, xh);

  // ---- 100 serially-dependent decoder steps ----
  for (int l = 0; l < LL; ++l) {
    k_gates_gemm<<<64, 256, 0, stream>>>(Wg, xh, gt, l, gates);
    k_lstm<<<(BB * DD) / 256, 256, 0, stream>>>(gates, bg, hbuf, cbuf, xh, concat);
    k_energy<<<dim3(TT / 8, BB), 256, 0, stream>>>(lfb, hbuf, energy);
    k_softmax_ctx<<<BB, 256, 0, stream>>>(energy, lfb, xh, concat,
                                          att_base + (size_t)l * BB * TT);
    k_out_gemm<<<1, 256, 0, stream>>>(concat, Wob, bout,
                                      raw_base + (size_t)l * BB * VV);
  }
}
